// lstm_36481452212721
// MI455X (gfx1250) — matrix operand order repaired and checked by reading, not compiled, not run on hardware
//
#include <hip/hip_runtime.h>
#include <hip/hip_bf16.h>

// Problem constants
#define TT 2048
#define BB 128
#define FF 24
#define HH 60
#define G4 240   // 4*H

typedef __attribute__((ext_vector_type(16))) _Float16 v16h;
typedef __attribute__((ext_vector_type(8)))  float    v8f;

union Frag16 { v16h v; uint4 q[2]; };

// A-fragment: 16x32 f16 tile, row-major source. rowp = &A[row0 + lane%16][k0], 16B aligned.
// ISA layout: lanes 0-15 hold K=0..7 (v0..3) and K=16..23 (v4..7); lanes 16-31 hold K=8..15 / K=24..31.
// Per lane this is two contiguous 16B chunks -> two b128 loads.
__device__ __forceinline__ v16h load_a16(const _Float16* rowp, int lane) {
  const uint4* q = (const uint4*)rowp;          // 16B chunks (8 f16 each)
  const int hi = (lane & 16) ? 1 : 0;           // second half of each 16-half group
  Frag16 f;
  f.q[0] = q[hi];
  f.q[1] = q[2 + hi];
  return f.v;
}

// B-fragment: 32x16 f16 tile (KxN) sourced from a row-major [N][K] weight matrix.
// colp = &W[n0 + lane%16][k0], 16B aligned. Lanes 0-15: K=0..7,16..23, lanes 16-31: K=8..15,24..31.
// Per lane: two 16B chunks, 32B apart -> two b128 loads.
__device__ __forceinline__ v16h load_b16(const _Float16* colp, int lane) {
  const uint4* q = (const uint4*)colp;
  const int o = (lane & 16) ? 1 : 0;
  Frag16 f;
  f.q[0] = q[o + 0];
  f.q[1] = q[o + 2];
  return f.v;
}

__device__ __forceinline__ float sigm(float x) { return 1.f / (1.f + __expf(-x)); }

// ---------------------------------------------------------------------------
// Conversion / init kernels
// ---------------------------------------------------------------------------
__global__ void cvt_x_kernel(const float* __restrict__ x, _Float16* __restrict__ x16) {
  int i = blockIdx.x * 256 + threadIdx.x;            // total TT*BB*32
  if (i >= TT * BB * 32) return;
  int f = i & 31;
  int rem = i >> 5;            // t*BB + b
  int b = rem & (BB - 1);
  int t = rem >> 7;
  x16[i] = (f < FF) ? (_Float16)x[((size_t)b * TT + t) * FF + f] : (_Float16)0.f;
}

__global__ void cvt_pad_kernel(const float* __restrict__ src, _Float16* __restrict__ dst,
                               int srows, int scols, int drows, int dcols) {
  int i = blockIdx.x * 256 + threadIdx.x;
  if (i >= drows * dcols) return;
  int r = i / dcols, c = i % dcols;
  dst[i] = (r < srows && c < scols) ? (_Float16)src[r * scols + c] : (_Float16)0.f;
}

__global__ void zero_u32_kernel(unsigned* __restrict__ p, int n) {
  int i = blockIdx.x * 256 + threadIdx.x;
  if (i < n) p[i] = 0u;
}

// ---------------------------------------------------------------------------
// Fused bidirectional LSTM scan.
// grid = 16 blocks: dir = bid&1, batch chunk of 16 rows = bid>>1. block = 512 (16 waves).
// Each wave owns ONE 16x16 N-tile (15 tiles, wave 15 idle in matmul); all weight
// B-fragments are packed into registers ONCE before the time loop.
// Per step: G[16,240] = X_t[16,KP] @ Wih^T + H[16,64] @ Whh^T  (WMMA f16->f32),
// gates in f32, h -> LDS (f16) and hout[(t*B+b)][128] col dir*60.
// LDS: Wih[240][KP] + Whh[240][64] + H[16][64] (f16) + G[16][240] (f32)
//   KP=32 -> 63,488 B ; KP=128 -> 109,568 B  (CDNA5: 320KB/WGP)
// ---------------------------------------------------------------------------
template <int KP>
__global__ void lstm_scan_kernel(const _Float16* __restrict__ xin,   // [TT*BB][KP] f16 (padded)
                                 const _Float16* __restrict__ wihF,  // [240][KP]
                                 const _Float16* __restrict__ whhF,  // [240][64]
                                 const float* __restrict__ bihF, const float* __restrict__ bhhF,
                                 const _Float16* __restrict__ wihR,
                                 const _Float16* __restrict__ whhR,
                                 const float* __restrict__ bihR, const float* __restrict__ bhhR,
                                 _Float16* __restrict__ hout)        // [TT*BB][128]
{
  extern __shared__ char smem[];
  _Float16* sWih = (_Float16*)smem;            // 240*KP
  _Float16* sWhh = sWih + 240 * KP;            // 240*64
  _Float16* sH   = sWhh + 240 * 64;            // 16*64
  float*    sG   = (float*)(sH + 16 * 64);     // 16*240

  const int tid  = threadIdx.x;
  const int lane = tid & 31;
  const int wave = tid >> 5;
  const int bid  = blockIdx.x;
  const int dir  = bid & 1;
  const int b0   = (bid >> 1) * 16;

  const _Float16* wih = dir ? wihR : wihF;
  const _Float16* whh = dir ? whhR : whhF;
  const float*    bih = dir ? bihR : bihF;
  const float*    bhh = dir ? bhhR : bhhF;

  // Stage weights to LDS; zero hidden state (cols 60..63 stay zero = K padding).
  for (int i = tid; i < 240 * KP / 2; i += 512) ((unsigned*)sWih)[i] = ((const unsigned*)wih)[i];
  for (int i = tid; i < 240 * 64 / 2; i += 512) ((unsigned*)sWhh)[i] = ((const unsigned*)whh)[i];
  for (int i = tid; i < 16 * 64 / 2; i += 512) ((unsigned*)sH)[i] = 0u;
  __syncthreads();

  // ---- hoist loop-invariant weight B-fragments into registers ----
  constexpr int KSX = KP / 32;                  // x-projection K-steps
  v16h bx[KSX];                                 // W_ih fragments for this wave's N-tile
  v16h bh[2];                                   // W_hh fragments
  const int n0 = wave * 16;
  if (wave < 15) {
    const _Float16* bc0 = sWih + (n0 + (lane & 15)) * KP;
#pragma unroll
    for (int k = 0; k < KSX; ++k) bx[k] = load_b16(bc0 + k * 32, lane);
    const _Float16* wc0 = sWhh + (n0 + (lane & 15)) * 64;
#pragma unroll
    for (int k = 0; k < 2; ++k) bh[k] = load_b16(wc0 + k * 32, lane);
  }

  // Gate ownership: e = tid*2+q -> local row bl = e>>6, unit j = e&63 (j<60 valid).
  float creg[2];
  float bi[2], bf[2], bg[2], bo[2];
#pragma unroll
  for (int q = 0; q < 2; ++q) {
    creg[q] = 0.f;
    int e = tid * 2 + q, j = e & 63;
    if (j < HH) {
      bi[q] = bih[j]        + bhh[j];
      bf[q] = bih[HH + j]   + bhh[HH + j];
      bg[q] = bih[2*HH + j] + bhh[2*HH + j];
      bo[q] = bih[3*HH + j] + bhh[3*HH + j];
    } else { bi[q] = bf[q] = bg[q] = bo[q] = 0.f; }
  }

  const int ml = lane & 15;                 // local batch row for A fragments
  const _Float16* hrow = sH + ml * 64;
  const int nn = n0 + (lane & 15);
  const int mb = (lane & 16) ? 8 : 0;

  for (int tt = 0; tt < TT; ++tt) {
    const int t = dir ? (TT - 1 - tt) : tt;
    const size_t rowbase = (size_t)t * BB + b0;

    // ---- matmul phase: one 16x16 tile per wave (waves 0..14) ----
    if (wave < 15) {
      v8f acc = {};
      const _Float16* arow = xin + (rowbase + ml) * KP;
#pragma unroll
      for (int k = 0; k < KSX; ++k) {
        v16h a = load_a16(arow + k * 32, lane);
        acc = __builtin_amdgcn_wmma_f32_16x16x32_f16(false, a, false, bx[k], (short)0, acc, false, false);
      }
#pragma unroll
      for (int k = 0; k < 2; ++k) {
        v16h a = load_a16(hrow + k * 32, lane);
        acc = __builtin_amdgcn_wmma_f32_16x16x32_f16(false, a, false, bh[k], (short)0, acc, false, false);
      }
#pragma unroll
      for (int r = 0; r < 8; ++r) sG[(mb + r) * G4 + nn] = acc[r];
    }

    // Prefetch next timestep's activation rows while gates compute.
    if (tt + 1 < TT) {
      const int tn = dir ? (t - 1) : (t + 1);
      const size_t nrow = (size_t)tn * BB + b0;
      constexpr int LPR = (KP * 2 + 127) / 128;     // 128B cachelines per row
      if (tid < 16 * LPR) {
        int r = tid / LPR, l = tid % LPR;
        __builtin_prefetch(xin + (nrow + r) * KP + l * 64, 0, 1);
      }
    }
    __syncthreads();

    // ---- gate phase: 2 elements per thread ----
#pragma unroll
    for (int q = 0; q < 2; ++q) {
      int e = tid * 2 + q;
      int bl = e >> 6, j = e & 63;
      if (j < HH) {
        const float* gr = sG + bl * G4;
        float iv = sigm(gr[j]          + bi[q]);
        float fv = sigm(gr[HH + j]     + bf[q]);
        float gv = tanhf(gr[2*HH + j]  + bg[q]);
        float ov = sigm(gr[3*HH + j]   + bo[q]);
        float c = fv * creg[q] + iv * gv;
        creg[q] = c;
        float h = ov * tanhf(c);
        sH[bl * 64 + j] = (_Float16)h;
        hout[(rowbase + bl) * 128 + dir * HH + j] = (_Float16)h;
      }
    }
    __syncthreads();
  }
}

// ---------------------------------------------------------------------------
// Attention scores: s[b,t] = tanh(out[t*B+b,:] @ W1^T + b1) . w2 + b2
// via WMMA GEMM [64 rows x 64 cols x K=128] per block, epilogue in LDS.
// ---------------------------------------------------------------------------
__global__ void attn_score_kernel(const _Float16* __restrict__ out16, // [TT*BB][128]
                                  const _Float16* __restrict__ w1p,   // [64][128] padded
                                  const float* __restrict__ b1,       // [60]
                                  const float* __restrict__ w2,       // [60]
                                  const float* __restrict__ b2,       // [1]
                                  float* __restrict__ sbuf)           // [BB][TT]
{
  __shared__ float sA[64 * 64];
  const int tid = threadIdx.x, lane = tid & 31, wave = tid >> 5;
  const size_t row0 = (size_t)blockIdx.x * 64;

  for (int tile = wave; tile < 16; tile += 8) {      // 4x4 tiles, 2 per wave
    const int mt = tile >> 2, nt = tile & 3;
    const _Float16* arow = out16 + (row0 + mt * 16 + (lane & 15)) * 128;
    const _Float16* bcol = w1p + (nt * 16 + (lane & 15)) * 128;
    v8f acc = {};
#pragma unroll
    for (int k0 = 0; k0 < 128; k0 += 32) {
      v16h a = load_a16(arow + k0, lane);
      v16h b = load_b16(bcol + k0, lane);
      acc = __builtin_amdgcn_wmma_f32_16x16x32_f16(false, a, false, b, (short)0, acc, false, false);
    }
    const int nn = nt * 16 + (lane & 15);
    const int mb = mt * 16 + ((lane & 16) ? 8 : 0);
#pragma unroll
    for (int r = 0; r < 8; ++r) sA[(mb + r) * 64 + nn] = acc[r];
  }
  __syncthreads();

  if (tid < 64) {
    float s = b2[0];
    const float* ar = sA + tid * 64;
    for (int k = 0; k < HH; ++k) s += tanhf(ar[k] + b1[k]) * w2[k];
    const size_t row = row0 + tid;               // = t*BB + b
    const int b = (int)(row & (BB - 1));
    const int t = (int)(row >> 7);
    sbuf[(size_t)b * TT + t] = s;
  }
}

// ---------------------------------------------------------------------------
// Softmax over T, attention pooling, fc + label head. One block per batch row.
// ---------------------------------------------------------------------------
__global__ void attn_pool_kernel(float* __restrict__ sbuf,            // [BB][TT] (overwritten w/ exp)
                                 const _Float16* __restrict__ out16,  // [TT*BB][128]
                                 const float* __restrict__ fcw, const float* __restrict__ fcb,
                                 const float* __restrict__ lw, const float* __restrict__ lb,
                                 float* __restrict__ outp)            // [BB]
{
  __shared__ float red[256];
  __shared__ float hid[120];
  __shared__ float fcv[10];
  __shared__ float sm[2];
  const int b = blockIdx.x, tid = threadIdx.x;
  float* srow = sbuf + (size_t)b * TT;

  float m = -1e30f;
  for (int t = tid; t < TT; t += 256) m = fmaxf(m, srow[t]);
  red[tid] = m; __syncthreads();
  for (int s = 128; s > 0; s >>= 1) { if (tid < s) red[tid] = fmaxf(red[tid], red[tid + s]); __syncthreads(); }
  if (tid == 0) sm[0] = red[0];
  __syncthreads();

  const float mx = sm[0];
  float acc = 0.f;
  for (int t = tid; t < TT; t += 256) { float e = __expf(srow[t] - mx); srow[t] = e; acc += e; }
  red[tid] = acc; __syncthreads();
  for (int s = 128; s > 0; s >>= 1) { if (tid < s) red[tid] += red[tid + s]; __syncthreads(); }
  if (tid == 0) sm[1] = red[0];
  __syncthreads();

  const float inv = 1.f / sm[1];
  if (tid < 120) {
    float h = 0.f;
    for (int t = 0; t < TT; ++t)
      h += srow[t] * (float)out16[((size_t)t * BB + b) * 128 + tid];
    hid[tid] = h * inv;
  }
  __syncthreads();
  if (tid < 10) {
    float f = fcb[tid];
    for (int j = 0; j < 120; ++j) f += hid[j] * fcw[tid * 120 + j];
    fcv[tid] = f;
  }
  __syncthreads();
  if (tid == 0) {
    float r = lb[0];
    for (int i = 0; i < 10; ++i) r += fcv[i] * lw[i];
    outp[b] = r;
  }
}

// ---------------------------------------------------------------------------
extern "C" void kernel_launch(void* const* d_in, const int* in_sizes, int n_in,
                              void* d_out, int out_size, void* d_ws, size_t ws_size,
                              hipStream_t stream) {
  (void)in_sizes; (void)n_in; (void)out_size; (void)ws_size;
  const float* x        = (const float*)d_in[0];
  const float* w_ih_l0  = (const float*)d_in[1];
  const float* w_hh_l0  = (const float*)d_in[2];
  const float* b_ih_l0  = (const float*)d_in[3];
  const float* b_hh_l0  = (const float*)d_in[4];
  const float* w_ih_l0r = (const float*)d_in[5];
  const float* w_hh_l0r = (const float*)d_in[6];
  const float* b_ih_l0r = (const float*)d_in[7];
  const float* b_hh_l0r = (const float*)d_in[8];
  const float* w_ih_l1  = (const float*)d_in[9];
  const float* w_hh_l1  = (const float*)d_in[10];
  const float* b_ih_l1  = (const float*)d_in[11];
  const float* b_hh_l1  = (const float*)d_in[12];
  const float* w_ih_l1r = (const float*)d_in[13];
  const float* w_hh_l1r = (const float*)d_in[14];
  const float* b_ih_l1r = (const float*)d_in[15];
  const float* b_hh_l1r = (const float*)d_in[16];
  const float* ws1_w    = (const float*)d_in[17];
  const float* ws1_b    = (const float*)d_in[18];
  const float* ws2_w    = (const float*)d_in[19];
  const float* ws2_b    = (const float*)d_in[20];
  const float* fc_w     = (const float*)d_in[21];
  const float* fc_b     = (const float*)d_in[22];
  const float* label_w  = (const float*)d_in[23];
  const float* label_b  = (const float*)d_in[24];

  char* ws = (char*)d_ws;
  size_t off = 0;
  auto alloc = [&](size_t bytes) -> char* {
    char* p = ws + off;
    off += (bytes + 255) & ~(size_t)255;
    return p;
  };
  _Float16* x16    = (_Float16*)alloc((size_t)TT * BB * 32 * 2);
  _Float16* h016   = (_Float16*)alloc((size_t)TT * BB * 128 * 2);
  _Float16* o116   = (_Float16*)alloc((size_t)TT * BB * 128 * 2);
  float*    sbuf   = (float*)alloc((size_t)BB * TT * 4);
  _Float16* wih0F  = (_Float16*)alloc(240 * 32 * 2);
  _Float16* wih0R  = (_Float16*)alloc(240 * 32 * 2);
  _Float16* whh0F  = (_Float16*)alloc(240 * 64 * 2);
  _Float16* whh0R  = (_Float16*)alloc(240 * 64 * 2);
  _Float16* wih1F  = (_Float16*)alloc(240 * 128 * 2);
  _Float16* wih1R  = (_Float16*)alloc(240 * 128 * 2);
  _Float16* whh1F  = (_Float16*)alloc(240 * 64 * 2);
  _Float16* whh1R  = (_Float16*)alloc(240 * 64 * 2);
  _Float16* w1p    = (_Float16*)alloc(64 * 128 * 2);

  // Conversions / padding to f16
  cvt_x_kernel<<<(TT * BB * 32 + 255) / 256, 256, 0, stream>>>(x, x16);
  cvt_pad_kernel<<<(240 * 32 + 255) / 256, 256, 0, stream>>>(w_ih_l0,  wih0F, 240, FF, 240, 32);
  cvt_pad_kernel<<<(240 * 32 + 255) / 256, 256, 0, stream>>>(w_ih_l0r, wih0R, 240, FF, 240, 32);
  cvt_pad_kernel<<<(240 * 64 + 255) / 256, 256, 0, stream>>>(w_hh_l0,  whh0F, 240, HH, 240, 64);
  cvt_pad_kernel<<<(240 * 64 + 255) / 256, 256, 0, stream>>>(w_hh_l0r, whh0R, 240, HH, 240, 64);
  cvt_pad_kernel<<<(240 * 128 + 255) / 256, 256, 0, stream>>>(w_ih_l1,  wih1F, 240, 120, 240, 128);
  cvt_pad_kernel<<<(240 * 128 + 255) / 256, 256, 0, stream>>>(w_ih_l1r, wih1R, 240, 120, 240, 128);
  cvt_pad_kernel<<<(240 * 64 + 255) / 256, 256, 0, stream>>>(w_hh_l1,  whh1F, 240, HH, 240, 64);
  cvt_pad_kernel<<<(240 * 64 + 255) / 256, 256, 0, stream>>>(w_hh_l1r, whh1R, 240, HH, 240, 64);
  cvt_pad_kernel<<<(64 * 128 + 255) / 256, 256, 0, stream>>>(ws1_w, w1p, HH, 120, 64, 128);
  // Zero padded activation buffers (K-padding columns must be 0)
  zero_u32_kernel<<<(TT * BB * 64 + 255) / 256, 256, 0, stream>>>((unsigned*)h016, TT * BB * 64);
  zero_u32_kernel<<<(TT * BB * 64 + 255) / 256, 256, 0, stream>>>((unsigned*)o116, TT * BB * 64);

  // Layer 0 (fwd+rev as blocks), then layer 1
  const size_t smem0 = ((size_t)240 * 32 + 240 * 64 + 16 * 64) * 2 + 16 * G4 * 4;
  lstm_scan_kernel<32><<<16, 512, smem0, stream>>>(x16, wih0F, whh0F, b_ih_l0, b_hh_l0,
                                                   wih0R, whh0R, b_ih_l0r, b_hh_l0r, h016);
  const size_t smem1 = ((size_t)240 * 128 + 240 * 64 + 16 * 64) * 2 + 16 * G4 * 4;
  lstm_scan_kernel<128><<<16, 512, smem1, stream>>>(h016, wih1F, whh1F, b_ih_l1, b_hh_l1,
                                                    wih1R, whh1R, b_ih_l1r, b_hh_l1r, o116);

  // Attention scores (WMMA), softmax + pooling + heads
  attn_score_kernel<<<TT * BB / 64, 256, 0, stream>>>(o116, w1p, ws1_b, ws2_w, ws2_b, sbuf);
  attn_pool_kernel<<<BB, 256, 0, stream>>>(sbuf, o116, fc_w, fc_b, label_w, label_b, (float*)d_out);
}
